// GroupedQueryAttention_7129645711713
// MI455X (gfx1250) — matrix operand order repaired and checked by reading, not compiled, not run on hardware
//
#include <hip/hip_runtime.h>
#include <hip/hip_bf16.h>

typedef unsigned short u16;
typedef unsigned int   u32;

typedef __attribute__((ext_vector_type(16))) __bf16 v16bf;
typedef __attribute__((ext_vector_type(8)))  float  v8f;

#define D_MODEL    768
#define N_HEADS    12
#define N_KV       4
#define D_HEAD     64
#define WINDOW     512
#define GLOBAL_TOK 64
#define BATCH      2
#define SEQ        2048
#define BT         (BATCH * SEQ)     // 4096 tokens
#define GROUPS     (N_HEADS / N_KV)  // 3

union BfFrag {
  v16bf v;
  u16   u[16];
  uint4 q[2];     // two 16-byte halves (A-layout fragments are 2 contiguous runs)
};

__device__ __forceinline__ u16 f2bf(float f) {
  u32 u = __float_as_uint(f);
  u32 r = u + 0x7FFFu + ((u >> 16) & 1u);   // round-to-nearest-even
  return (u16)(r >> 16);
}

__device__ __forceinline__ v8f wmma_bf16(v16bf a, v16bf b, v8f c) {
  // D = A(16x32 bf16) * B(32x16 bf16) + C(16x16 f32)
  return __builtin_amdgcn_wmma_f32_16x16x32_bf16(false, a, false, b, (short)0, c,
                                                 false, false);
}

// ---------------------------------------------------------------------------
// f32 -> bf16 convert
// ---------------------------------------------------------------------------
__global__ __launch_bounds__(256) void k_f32_to_bf16(const float* __restrict__ in,
                                                     u16* __restrict__ out, int n) {
  int i = blockIdx.x * 256 + threadIdx.x;
  if (i < n) out[i] = f2bf(in[i]);
}

// ---------------------------------------------------------------------------
// bf16 GEMM: C[M,N] f32 = A[M,K] * B[K,N], row-major operands.
// Block = 256 thr (8 waves). Tile: 128 rows x 16 cols, K-step 32.
// A tile staged row-major (b128 in/out); B tile staged COLUMN-major so each
// lane's WMMA B-fragment is two contiguous ds_load_b128s.
// Requires M%128==0, N%16==0 (N%8==0 for loads), K%32==0.
// ---------------------------------------------------------------------------
__global__ __launch_bounds__(256) void k_gemm_bf16(const u16* __restrict__ A,
                                                   const u16* __restrict__ B,
                                                   float* __restrict__ C,
                                                   int M, int N, int K) {
  __shared__ uint4 aT4[128 * 4];   // 128 x 32 bf16, row-major
  __shared__ uint4 bT4[16 * 4];    // 16 cols x 32 k, column-major [n][k]
  u16* aTile = (u16*)aT4;
  u16* bCM   = (u16*)bT4;

  const int tid  = threadIdx.x;
  const int wave = tid >> 5;
  const int lane = tid & 31;
  const int m0   = blockIdx.y * 128;
  const int n0   = blockIdx.x * 16;

  v8f acc = {};

  const uint4* Aq = (const uint4*)A;
  const int lda4 = K >> 3;   // uint4 per A row

  for (int k0 = 0; k0 < K; k0 += 32) {
    // ---- A tile: 512 uint4, 2 per thread (b128 load + b128 LDS store) ----
#pragma unroll
    for (int j = 0; j < 2; ++j) {
      int idx = tid + j * 256;
      int r = idx >> 2, qq = idx & 3;
      aT4[idx] = Aq[(size_t)(m0 + r) * lda4 + (k0 >> 3) + qq];
    }
    // ---- B tile: 32x16, transpose into column-major LDS [n][k] ----
#pragma unroll
    for (int j = 0; j < 2; ++j) {
      int idx = tid + j * 256;       // 512 elements
      int kk = idx >> 4, nn = idx & 15;
      bCM[nn * 32 + kk] = B[(size_t)(k0 + kk) * N + n0 + nn];
    }
    if (k0 + 32 < K)   // prefetch next A tile chunk for this thread
      __builtin_prefetch(&Aq[(size_t)(m0 + (tid >> 2)) * lda4 + ((k0 + 32) >> 3)], 0, 1);
    __syncthreads();

    // ---- fragments: 2 x b128 each ----
    BfFrag af, bf;
    const int mloc = wave * 16 + (lane & 15);
    const int hi8  = (lane >> 4) << 3;    // 0 or 8
    af.q[0] = *(const uint4*)&aTile[mloc * 32 + hi8];
    af.q[1] = *(const uint4*)&aTile[mloc * 32 + 16 + hi8];
    const int hk = (lane >> 4) << 4;      // 0 or 16
    bf.q[0] = *(const uint4*)&bCM[(lane & 15) * 32 + hi8];
    bf.q[1] = *(const uint4*)&bCM[(lane & 15) * 32 + 16 + hi8];
    acc = wmma_bf16(af.v, bf.v, acc);
    __syncthreads();
  }

#pragma unroll
  for (int r = 0; r < 8; ++r) {
    int mm = m0 + wave * 16 + r + ((lane >> 4) << 3);
    int nn = n0 + (lane & 15);
    C[(size_t)mm * N + nn] = acc[r];
  }
}

// ---------------------------------------------------------------------------
// RoPE + scatter. One block per token.
//  qf[BT,768] -> qb16 [B,H,T,64]    (bf16, roped)
//  kf[BT,256] -> kb16 [B,KV,T,64]   (bf16, roped)  + new_k f32 (d_out)
//  vf[BT,256] -> vbT  [B,KV,64,T]   (bf16, TRANSPOSED for PV fragments)
//                                   + new_v f32 (d_out, [B,KV,T,64])
// ---------------------------------------------------------------------------
__global__ __launch_bounds__(256) void k_rope_scatter(const float* __restrict__ qf,
                                                      const float* __restrict__ kf,
                                                      const float* __restrict__ vf,
                                                      u16* __restrict__ qb,
                                                      u16* __restrict__ kb,
                                                      u16* __restrict__ vbT,
                                                      float* __restrict__ new_k,
                                                      float* __restrict__ new_v) {
  const int token = blockIdx.x;          // 0..BT-1
  const int b = token / SEQ;
  const int t = token % SEQ;
  const int tid = threadIdx.x;

  // Q: N_HEADS * 32 rotation pairs
  for (int idx = tid; idx < N_HEADS * 32; idx += 256) {
    int hh = idx >> 5, d = idx & 31;
    float inv = __powf(10000.0f, -(float)d * (1.0f / 32.0f));
    float s, c;
    __sincosf((float)t * inv, &s, &c);
    float x0 = qf[(size_t)token * D_MODEL + hh * 64 + d];
    float x1 = qf[(size_t)token * D_MODEL + hh * 64 + d + 32];
    size_t o = ((size_t)(b * N_HEADS + hh) * SEQ + t) * 64 + d;
    qb[o]      = f2bf(x0 * c - x1 * s);
    qb[o + 32] = f2bf(x1 * c + x0 * s);
  }
  // K: N_KV * 32 rotation pairs
  for (int idx = tid; idx < N_KV * 32; idx += 256) {
    int kv = idx >> 5, d = idx & 31;
    float inv = __powf(10000.0f, -(float)d * (1.0f / 32.0f));
    float s, c;
    __sincosf((float)t * inv, &s, &c);
    float x0 = kf[(size_t)token * (N_KV * 64) + kv * 64 + d];
    float x1 = kf[(size_t)token * (N_KV * 64) + kv * 64 + d + 32];
    float r0 = x0 * c - x1 * s;
    float r1 = x1 * c + x0 * s;
    size_t o = ((size_t)(b * N_KV + kv) * SEQ + t) * 64 + d;
    kb[o] = f2bf(r0);  kb[o + 32] = f2bf(r1);
    new_k[o] = r0;     new_k[o + 32] = r1;
  }
  // V: copy (normal layout to d_out, transposed bf16 for attention)
  for (int idx = tid; idx < N_KV * 64; idx += 256) {
    int kv = idx >> 6, d = idx & 63;
    float x = vf[(size_t)token * (N_KV * 64) + kv * 64 + d];
    new_v[((size_t)(b * N_KV + kv) * SEQ + t) * 64 + d] = x;
    vbT[((size_t)(b * N_KV + kv) * D_HEAD + d) * SEQ + t] = f2bf(x);
  }
}

// ---------------------------------------------------------------------------
// Flash attention. One wave per 16-query tile; 8 independent waves per block.
// Key tiles of 32; skip tiles fully outside [0,GLOBAL) U (q0-WINDOW, q0+15].
// K and V fragments are single contiguous 32-byte vector loads per lane.
// ---------------------------------------------------------------------------
__global__ __launch_bounds__(256) void k_attn(const u16* __restrict__ qb,
                                              const u16* __restrict__ kb,
                                              const u16* __restrict__ vbT,
                                              u16* __restrict__ ab) {
  __shared__ u16 plds[8][16 * 32];   // per-wave P staging (per-wave LDS is in-order)

  const int tid  = threadIdx.x;
  const int wave = tid >> 5;
  const int lane = tid & 31;
  const int w    = blockIdx.x * 8 + wave;
  const int tilesPerB = N_HEADS * (SEQ / 16);
  if (w >= BATCH * tilesPerB) return;

  const int b   = w / tilesPerB;
  const int rem = w % tilesPerB;
  const int h   = rem / (SEQ / 16);
  const int q0  = (rem % (SEQ / 16)) * 16;
  const int kvh = h / GROUPS;

  const u16* qbase = qb  + ((size_t)(b * N_HEADS + h) * SEQ) * D_HEAD;
  const u16* kbase = kb  + ((size_t)(b * N_KV + kvh) * SEQ) * D_HEAD;
  const u16* vtb   = vbT + ((size_t)(b * N_KV + kvh) * D_HEAD) * SEQ;

  // Q fragments (A layout, 16x32 each) for d-chunks 0/1: 2 x b128 each
  BfFrag qfr[2];
  {
    const int mrow = q0 + (lane & 15);
    const int hi8  = (lane >> 4) << 4;
#pragma unroll
    for (int dc = 0; dc < 2; ++dc) {
      qfr[dc].q[0] = *(const uint4*)&qbase[(size_t)mrow * D_HEAD + dc * 32 + hi8];
      qfr[dc].q[1] = *(const uint4*)&qbase[(size_t)mrow * D_HEAD + dc * 32 + 8 + hi8];
    }
  }

  v8f zero = {};
  v8f o[4] = {zero, zero, zero, zero};
  float rmax[8], rsum[8];
#pragma unroll
  for (int r = 0; r < 8; ++r) { rmax[r] = -3.0e38f; rsum[r] = 0.0f; }

  const int qEnd = q0 + 15;
  for (int j0 = 0; j0 <= qEnd; j0 += 32) {
    if (j0 >= GLOBAL_TOK && (j0 + 31) < (q0 - WINDOW + 1)) continue;  // fully masked

    // prefetch next key tile (speculative; skipped-range prefetch is harmless)
    if (j0 + 32 <= qEnd) {
      __builtin_prefetch(&kbase[(size_t)(j0 + 32 + (lane & 31)) * D_HEAD], 0, 1);
      __builtin_prefetch(&vtb[(size_t)(lane & 31) * SEQ + j0 + 32], 0, 1);
    }

    // ---- S = Q K^T for two 16-key subtiles ----
    v8f s0 = zero, s1 = zero;
#pragma unroll
    for (int dc = 0; dc < 2; ++dc) {
      // B frag: lane holds key = base + lane%16, d = dc*32 + (lane/16)*16 + e
      // => 16 contiguous bf16 per lane: one 32-byte vector load.
      BfFrag kf0, kf1;
      kf0.v = *(const v16bf*)&kbase[(size_t)(j0 + (lane & 15)) * D_HEAD +
                                    dc * 32 + ((lane >> 4) << 4)];
      kf1.v = *(const v16bf*)&kbase[(size_t)(j0 + 16 + (lane & 15)) * D_HEAD +
                                    dc * 32 + ((lane >> 4) << 4)];
      s0 = wmma_bf16(qfr[dc].v, kf0.v, s0);
      s1 = wmma_bf16(qfr[dc].v, kf1.v, s1);
    }

    // ---- mask + online softmax (rows spread over 16 lanes, reduce via shfl) ----
    const float scale = 0.125f;   // 1/sqrt(64)
#pragma unroll
    for (int r = 0; r < 8; ++r) {
      int i  = q0 + r + ((lane >> 4) << 3);
      int jA = j0 + (lane & 15);
      int jB = jA + 16;
      bool okA = (jA <= i) && ((jA > i - WINDOW) || (jA < GLOBAL_TOK) || (i < GLOBAL_TOK));
      bool okB = (jB <= i) && ((jB > i - WINDOW) || (jB < GLOBAL_TOK) || (i < GLOBAL_TOK));
      float a0 = okA ? s0[r] * scale : -3.0e38f;
      float a1 = okB ? s1[r] * scale : -3.0e38f;

      float tmax = fmaxf(a0, a1);
      for (int off = 8; off >= 1; off >>= 1)
        tmax = fmaxf(tmax, __shfl_xor(tmax, off, 16));
      float nm = fmaxf(rmax[r], tmax);
      float fr = __expf(rmax[r] - nm);
      rmax[r] = nm;
      rsum[r] *= fr;
      o[0][r] *= fr; o[1][r] *= fr; o[2][r] *= fr; o[3][r] *= fr;

      float e0 = __expf(a0 - nm);
      float e1 = __expf(a1 - nm);
      float ps = e0 + e1;
      for (int off = 8; off >= 1; off >>= 1)
        ps += __shfl_xor(ps, off, 16);
      rsum[r] += ps;

      int mm = r + ((lane >> 4) << 3);
      plds[wave][mm * 32 + (lane & 15)]      = f2bf(e0);
      plds[wave][mm * 32 + 16 + (lane & 15)] = f2bf(e1);
    }

    // ---- P (A layout 16x32) from this wave's LDS region: 2 x ds_load_b128 ----
    BfFrag pf;
    {
      const int mm  = lane & 15;
      const int hi8 = (lane >> 4) << 4;
      pf.q[0] = *(const uint4*)&plds[wave][mm * 32 + hi8];
      pf.q[1] = *(const uint4*)&plds[wave][mm * 32 + 8 + hi8];
    }

    // ---- O += P * V  (four 16-wide d chunks; V^T => contiguous loads) ----
#pragma unroll
    for (int c = 0; c < 4; ++c) {
      BfFrag vfr;
      vfr.v = *(const v16bf*)&vtb[(size_t)(c * 16 + (lane & 15)) * SEQ +
                                  j0 + ((lane >> 4) << 4)];
      o[c] = wmma_bf16(pf.v, vfr.v, o[c]);
    }
  }

  // ---- normalize + store to ab [B,T,H*D] as bf16 ----
#pragma unroll
  for (int r = 0; r < 8; ++r) {
    float inv = 1.0f / rsum[r];
    int t = q0 + r + ((lane >> 4) << 3);
#pragma unroll
    for (int c = 0; c < 4; ++c) {
      int col = h * D_HEAD + c * 16 + (lane & 15);
      ab[((size_t)(b * SEQ + t)) * D_MODEL + col] = f2bf(o[c][r] * inv);
    }
  }
}

// ---------------------------------------------------------------------------
// Host-side orchestration
// ---------------------------------------------------------------------------
static size_t alignup(size_t v) { return (v + 255) & ~(size_t)255; }

extern "C" void kernel_launch(void* const* d_in, const int* in_sizes, int n_in,
                              void* d_out, int out_size, void* d_ws, size_t ws_size,
                              hipStream_t stream) {
  (void)in_sizes; (void)n_in; (void)out_size; (void)ws_size;

  const float* x  = (const float*)d_in[0];
  const float* wq = (const float*)d_in[1];
  const float* wk = (const float*)d_in[2];
  const float* wv = (const float*)d_in[3];
  const float* wo = (const float*)d_in[4];

  float* out   = (float*)d_out;                                   // [B,T,768]
  float* new_k = out + (size_t)BT * D_MODEL;                      // [B,KV,T,64]
  float* new_v = new_k + (size_t)BATCH * N_KV * SEQ * D_HEAD;     // [B,KV,T,64]

  char* ws = (char*)d_ws;
  auto take = [&](size_t bytes) { char* p = ws; ws += alignup(bytes); return p; };

  u16*   xb  = (u16*)take((size_t)BT * D_MODEL * 2);
  u16*   wqb = (u16*)take((size_t)D_MODEL * (N_HEADS * D_HEAD) * 2);
  u16*   wkb = (u16*)take((size_t)D_MODEL * (N_KV * D_HEAD) * 2);
  u16*   wvb = (u16*)take((size_t)D_MODEL * (N_KV * D_HEAD) * 2);
  u16*   wob = (u16*)take((size_t)(N_HEADS * D_HEAD) * D_MODEL * 2);
  float* qf  = (float*)take((size_t)BT * (N_HEADS * D_HEAD) * 4);
  float* kf  = (float*)take((size_t)BT * (N_KV * D_HEAD) * 4);
  float* vf  = (float*)take((size_t)BT * (N_KV * D_HEAD) * 4);
  u16*   qb  = (u16*)take((size_t)BATCH * N_HEADS * SEQ * D_HEAD * 2);
  u16*   kb  = (u16*)take((size_t)BATCH * N_KV * SEQ * D_HEAD * 2);
  u16*   vbT = (u16*)take((size_t)BATCH * N_KV * D_HEAD * SEQ * 2);
  u16*   ab  = (u16*)take((size_t)BT * D_MODEL * 2);

  auto conv = [&](const float* src, u16* dst, int n) {
    k_f32_to_bf16<<<dim3((n + 255) / 256), dim3(256), 0, stream>>>(src, dst, n);
  };
  conv(x,  xb,  BT * D_MODEL);
  conv(wq, wqb, D_MODEL * N_HEADS * D_HEAD);
  conv(wk, wkb, D_MODEL * N_KV * D_HEAD);
  conv(wv, wvb, D_MODEL * N_KV * D_HEAD);
  conv(wo, wob, N_HEADS * D_HEAD * D_MODEL);

  // QKV projections (WMMA GEMMs)
  k_gemm_bf16<<<dim3((N_HEADS * D_HEAD) / 16, BT / 128), dim3(256), 0, stream>>>(
      xb, wqb, qf, BT, N_HEADS * D_HEAD, D_MODEL);
  k_gemm_bf16<<<dim3((N_KV * D_HEAD) / 16, BT / 128), dim3(256), 0, stream>>>(
      xb, wkb, kf, BT, N_KV * D_HEAD, D_MODEL);
  k_gemm_bf16<<<dim3((N_KV * D_HEAD) / 16, BT / 128), dim3(256), 0, stream>>>(
      xb, wvb, vf, BT, N_KV * D_HEAD, D_MODEL);

  // RoPE + kv-cache outputs + bf16 scatter for attention
  k_rope_scatter<<<dim3(BT), dim3(256), 0, stream>>>(qf, kf, vf, qb, kb, vbT,
                                                     new_k, new_v);

  // Flash attention: 3072 q-tiles, 8 waves/block
  k_attn<<<dim3((BATCH * N_HEADS * (SEQ / 16)) / 8), dim3(256), 0, stream>>>(
      qb, kb, vbT, ab);

  // Output projection straight into d_out
  k_gemm_bf16<<<dim3(D_MODEL / 16, BT / 128), dim3(256), 0, stream>>>(
      ab, wob, out, BT, D_MODEL, D_MODEL);
}
